// E82SelfGateCell_57097295233705
// MI455X (gfx1250) — compile-verified, matrix operand order checked by reading, not run on hardware
//
#include <hip/hip_runtime.h>
#include <hip/hip_bf16.h>

// ---------------------------------------------------------------------------
// E82 SelfGateCell for MI455X (gfx1250, wave32)
//   Kernel 0: W (256x1024 f32) -> bf16 once (0.5 MB, L2-resident for GEMM)
//   Kernel 1: proj = x @ W^T via v_wmma_f32_16x16x32_bf16, f32 accumulate
//             A converted in-register with v_perm_b32 packing (1.5 VALU/elem)
//   Kernel 2: sequential scan, 1 block/batch, state in regs, 2 barriers/step,
//             proj row staged via GLOBAL_LOAD_ASYNC_TO_LDS (if available)
// ---------------------------------------------------------------------------

#define DIM     1024
#define NS      64
#define NPROJ   256          // 4 * NS
#define TSTEPS  2048
#define BATCH   32

typedef __attribute__((ext_vector_type(16))) __bf16    v16bf;
typedef __attribute__((ext_vector_type(8)))  float     v8f;
typedef __attribute__((ext_vector_type(8)))  unsigned  v8u;

#define AS1 __attribute__((address_space(1)))
#define AS3 __attribute__((address_space(3)))

#if defined(__has_builtin)
#if __has_builtin(__builtin_amdgcn_global_load_async_to_lds_b32) && \
    __has_builtin(__builtin_amdgcn_s_wait_asynccnt)
#define HAVE_ASYNC_LDS 1
#endif
#endif

// Pack two f32 into one dword of two bf16 (round-half-up) with one v_perm_b32.
static __device__ __forceinline__ unsigned pack2(float f0, float f1) {
  unsigned u0 = __builtin_bit_cast(unsigned, f0) + 0x8000u;
  unsigned u1 = __builtin_bit_cast(unsigned, f1) + 0x8000u;
  // bytes 0-3 select from src1(u0), 4-7 from src0(u1):
  // result = {u1[31:24], u1[23:16], u0[31:24], u0[23:16]} = bf16(f0) | bf16(f1)<<16
  return __builtin_amdgcn_perm(u1, u0, 0x07060302u);
}

// ---------------------------------------------------------------------------
// Kernel 0: convert W (f32) to bf16 bits, 4 floats / thread.
// ---------------------------------------------------------------------------
__global__ __launch_bounds__(256)
void e82_cvt_w(const float* __restrict__ W, unsigned* __restrict__ Wb) {
  const int idx = blockIdx.x * 256 + threadIdx.x;   // index of float4
  const float4 f = ((const float4*)W)[idx];
  uint2 p;
  p.x = pack2(f.x, f.y);
  p.y = pack2(f.z, f.w);
  ((uint2*)Wb)[idx] = p;
}

// ---------------------------------------------------------------------------
// Kernel 1: proj[(t*B+b)][n] = sum_d x[(t*B+b)][d] * W[n][d]
// Block = 16 M-rows x 256 N-cols; 8 waves, wave w owns N-tiles 32w, 32w+16.
// A frag (16x32 bf16): lane L -> M=L%16, half=L/16; elems 0..7 hold
// K = kb*32+8*half+(0..7); elems 8..15 hold K = kb*32+16+8*half+(0..7).
// B frag (32x16): lane L -> N=L%16, same K pattern; B[k][n]=W[n][k], so each
// lane streams contiguous bf16 runs of one pre-converted W row (pure b128).
// ---------------------------------------------------------------------------
__global__ __launch_bounds__(256)
void e82_proj_gemm(const float* __restrict__ x,
                   const unsigned short* __restrict__ Wb,   // bf16 bits
                   float* __restrict__ proj) {
  const int lane = threadIdx.x & 31;
  const int wave = threadIdx.x >> 5;
  const int m0   = blockIdx.x << 4;
  const int n0   = wave << 5;
  const int half = lane >> 4;
  const int lidx = lane & 15;            // M for A-frag, N for B-frag

  const float*          __restrict__ arow  = x  + (size_t)(m0 + lidx) * DIM;
  const unsigned short* __restrict__ w0row = Wb + (size_t)(n0 + lidx) * DIM;
  const unsigned short* __restrict__ w1row = Wb + (size_t)(n0 + 16 + lidx) * DIM;

  v8f c0 = {};
  v8f c1 = {};

  #pragma unroll 4
  for (int kb = 0; kb < DIM / 32; ++kb) {
    const int k0 = (kb << 5) + (half << 3);

    // ---- A fragment: 4 x b128 f32 loads, perm-pack to bf16 ----
    float4 aA = *(const float4*)(arow + k0);
    float4 aB = *(const float4*)(arow + k0 + 4);
    float4 aC = *(const float4*)(arow + k0 + 16);
    float4 aD = *(const float4*)(arow + k0 + 20);
    v8u ap;
    ap[0] = pack2(aA.x, aA.y); ap[1] = pack2(aA.z, aA.w);
    ap[2] = pack2(aB.x, aB.y); ap[3] = pack2(aB.z, aB.w);
    ap[4] = pack2(aC.x, aC.y); ap[5] = pack2(aC.z, aC.w);
    ap[6] = pack2(aD.x, aD.y); ap[7] = pack2(aD.z, aD.w);
    const v16bf a = __builtin_bit_cast(v16bf, ap);

    // ---- B fragments: pre-converted bf16, 2 x b128 each, zero VALU ----
    uint4 b0lo = *(const uint4*)(w0row + k0);
    uint4 b0hi = *(const uint4*)(w0row + k0 + 16);
    v8u bp0;
    bp0[0] = b0lo.x; bp0[1] = b0lo.y; bp0[2] = b0lo.z; bp0[3] = b0lo.w;
    bp0[4] = b0hi.x; bp0[5] = b0hi.y; bp0[6] = b0hi.z; bp0[7] = b0hi.w;
    const v16bf b0 = __builtin_bit_cast(v16bf, bp0);

    uint4 b1lo = *(const uint4*)(w1row + k0);
    uint4 b1hi = *(const uint4*)(w1row + k0 + 16);
    v8u bp1;
    bp1[0] = b1lo.x; bp1[1] = b1lo.y; bp1[2] = b1lo.z; bp1[3] = b1lo.w;
    bp1[4] = b1hi.x; bp1[5] = b1hi.y; bp1[6] = b1hi.z; bp1[7] = b1hi.w;
    const v16bf b1 = __builtin_bit_cast(v16bf, bp1);

    c0 = __builtin_amdgcn_wmma_f32_16x16x32_bf16(false, a, false, b0,
                                                 (short)0, c0, false, false);
    c1 = __builtin_amdgcn_wmma_f32_16x16x32_bf16(false, a, false, b1,
                                                 (short)0, c1, false, false);
  }

  // store: lane -> col N = lidx; VGPR r -> row M = r + 8*half
  #pragma unroll
  for (int r = 0; r < 8; ++r) {
    const int row = m0 + r + (half << 3);
    proj[(size_t)row * NPROJ + n0 + lidx]      = c0[r];
    proj[(size_t)row * NPROJ + n0 + 16 + lidx] = c1[r];
  }
}

// ---------------------------------------------------------------------------
// Kernel 2: scan.  One block per batch, 512 threads (16 waves).
// Thread t owns S[i][jc..jc+7], i = t/8, jc = (t%8)*8.  Row reductions are
// width-8 shuffles (inside one wave32).  k/m norms computed redundantly per
// wave (full-wave shuffle reduce) and applied as scalar fix-ups AFTER the
// dot-product reductions -> only 2 barriers per step.
// ---------------------------------------------------------------------------
__global__ __launch_bounds__(512)
void e82_scan(const float* __restrict__ proj,
              const float* __restrict__ S0,
              const float* __restrict__ alpha_p,
              float* __restrict__ out,     // (T, B, 64)
              float* __restrict__ Sfin) {  // (B, 64, 64)
  __shared__ float pk[NPROJ];  // raw proj row: [0:64)k [64:128)v [128:192)q [192:256)m

  const int b    = blockIdx.x;
  const int tid  = threadIdx.x;
  const int lane = tid & 31;
  const int i    = tid >> 3;          // state row 0..63
  const int jc   = (tid & 7) << 3;    // state col chunk
  const float alpha = alpha_p[0];

  float S[8];
  #pragma unroll
  for (int j = 0; j < 8; ++j)
    S[j] = S0[(size_t)b * NS * NS + (size_t)i * NS + jc + j];

  for (int t = 0; t < TSTEPS; ++t) {
    const size_t base = ((size_t)t * BATCH + b) * NPROJ;

    __syncthreads();                    // prev-step pk reads complete
#ifdef HAVE_ASYNC_LDS
    if (tid < NPROJ) {
      AS1 int* g = (AS1 int*)(uintptr_t)(proj + base + tid);
      AS3 int* l = (AS3 int*)(unsigned)(uintptr_t)&pk[tid];
      __builtin_amdgcn_global_load_async_to_lds_b32(g, l, 0, 0);
    }
    __builtin_amdgcn_s_wait_asynccnt(0);
#else
    if (tid < NPROJ) pk[tid] = proj[base + tid];
#endif
    if (tid < NPROJ && t + 1 < TSTEPS)  // warm caches for next step
      __builtin_prefetch(proj + base + (size_t)BATCH * NPROJ + tid, 0, 1);
    __syncthreads();

    // per-wave redundant norms of k and m (raw values stay in LDS)
    float ka = pk[lane],       kb_ = pk[lane + 32];
    float ma = pk[192 + lane], mb_ = pk[192 + lane + 32];
    float ssk = ka * ka + kb_ * kb_;
    float ssm = ma * ma + mb_ * mb_;
    #pragma unroll
    for (int d = 16; d >= 1; d >>= 1) {
      ssk += __shfl_xor(ssk, d, 32);
      ssm += __shfl_xor(ssm, d, 32);
    }
    const float inv_k = 1.0f / (sqrtf(ssk) + 1e-6f);
    const float inv_m = 1.0f / (sqrtf(ssm) + 1e-6f);

    // Sm = S @ m_hat ; s_ret = S @ k_hat  (scale after reduce)
    float pm = 0.0f, pr = 0.0f;
    #pragma unroll
    for (int j = 0; j < 8; ++j) {
      pm += S[j] * pk[192 + jc + j];
      pr += S[j] * pk[jc + j];
    }
    #pragma unroll
    for (int d = 1; d < 8; d <<= 1) {
      pm += __shfl_xor(pm, d, 8);
      pr += __shfl_xor(pr, d, 8);
    }
    const float Sm     = pm * inv_m;
    const float sdelta = pk[64 + i] - pr * inv_k;   // v[i] - s_retrieved[i]

    // gate + state update + Sq partial
    float pq = 0.0f;
    #pragma unroll
    for (int j = 0; j < 8; ++j) {
      const float kj = pk[jc + j] * inv_k;
      const float gl = Sm * kj + alpha * S[j];
      const float g  = 1.0f / (1.0f + __expf(-gl));
      const float sn = g * S[j] + sdelta * kj;
      S[j] = sn;
      pq  += sn * pk[128 + jc + j];
    }
    #pragma unroll
    for (int d = 1; d < 8; d <<= 1) pq += __shfl_xor(pq, d, 8);

    if ((tid & 7) == 0) {
      const float sig = 1.0f / (1.0f + __expf(-pq));
      // out = Sq * silu(Sq) = Sq^2 * sigmoid(Sq)
      out[(size_t)t * (BATCH * NS) + (size_t)b * NS + i] = pq * pq * sig;
    }
  }

  #pragma unroll
  for (int j = 0; j < 8; ++j)
    Sfin[(size_t)b * NS * NS + (size_t)i * NS + jc + j] = S[j];
}

// ---------------------------------------------------------------------------
extern "C" void kernel_launch(void* const* d_in, const int* in_sizes, int n_in,
                              void* d_out, int out_size, void* d_ws, size_t ws_size,
                              hipStream_t stream) {
  (void)in_sizes; (void)n_in; (void)out_size; (void)ws_size;

  const float* x     = (const float*)d_in[0];  // (T, B, DIM)
  const float* S0    = (const float*)d_in[1];  // (B, 64, 64)
  const float* W     = (const float*)d_in[2];  // (256, DIM)
  const float* alpha = (const float*)d_in[3];  // scalar

  // workspace layout: [Wb bf16: 512 KB][proj f32: 64 MB]
  unsigned*       Wb_u  = (unsigned*)d_ws;
  unsigned short* Wb    = (unsigned short*)d_ws;
  float*          proj  = (float*)((char*)d_ws + (size_t)NPROJ * DIM * 2);

  float* out  = (float*)d_out;                 // (T, B, 64)
  float* Sfin = out + (size_t)TSTEPS * BATCH * NS;

  e82_cvt_w<<<dim3((NPROJ * DIM / 4) / 256), dim3(256), 0, stream>>>(W, Wb_u);

  const int Mtiles = (TSTEPS * BATCH) / 16;    // 4096 blocks
  e82_proj_gemm<<<dim3(Mtiles), dim3(256), 0, stream>>>(x, Wb, proj);
  e82_scan<<<dim3(BATCH), 512, 0, stream>>>(proj, S0, alpha, out, Sfin);
}
